// MultiHeadAttention_56375740727510
// MI455X (gfx1250) — matrix operand order repaired and checked by reading, not compiled, not run on hardware
//
#include <hip/hip_runtime.h>
#include <hip/hip_bf16.h>

#define BB  4
#define TT  2048
#define DD  1024
#define HH  16
#define DHH 64

typedef _Float16 v16h __attribute__((ext_vector_type(16)));
typedef _Float16 v8h  __attribute__((ext_vector_type(8)));
typedef float    v8f  __attribute__((ext_vector_type(8)));

__device__ __forceinline__ v8f fzero() {
    v8f z;
#pragma unroll
    for (int i = 0; i < 8; ++i) z[i] = 0.0f;
    return z;
}

__device__ __forceinline__ v16h frag_rowK(const _Float16* base, int row, int ld,
                                          int kA, int kB) {
    v8h lo = *reinterpret_cast<const v8h*>(base + (size_t)row * ld + kA);
    v8h hi = *reinterpret_cast<const v8h*>(base + (size_t)row * ld + kB);
    v16h f;
#pragma unroll
    for (int i = 0; i < 8; ++i) { f[i] = lo[i]; f[8 + i] = hi[i]; }
    return f;
}

// A fragment (16x32 f16): lane holds row = caller's row, K pairs per ISA layout
__device__ __forceinline__ v16h fragA(const _Float16* base, int row, int ld,
                                      int kbase, int ksel) {
    return frag_rowK(base, row, ld, kbase + 8 * ksel, kbase + 16 + 8 * ksel);
}
// B fragment (32x16 f16): lane holds column = caller's row index in (N,K) storage
__device__ __forceinline__ v16h fragB(const _Float16* base, int row, int ld,
                                      int kbase, int ksel) {
    return frag_rowK(base, row, ld, kbase + 8 * ksel, kbase + 16 + 8 * ksel);
}

__device__ __forceinline__ v8f wmma_f16(v16h a, v16h b, v8f c) {
    return __builtin_amdgcn_wmma_f32_16x16x32_f16(false, a, false, b,
                                                  (short)0, c, false, false);
}

// ---------------- f32 -> f16 conversion ----------------
__global__ void cvt_f16_kernel(const float* __restrict__ in,
                               _Float16* __restrict__ out, int n) {
    int i = blockIdx.x * blockDim.x + threadIdx.x;
    if (i < n) out[i] = (_Float16)in[i];
}

// ---------------- fused QKV projection + RoPE ----------------
// grid: (64, 16, 3), block 128.  wave -> 32 tokens x 64 feats (one head)
__global__ __launch_bounds__(128) void qkv_proj_kernel(
    const _Float16* __restrict__ xh,
    const _Float16* __restrict__ Wqh, const _Float16* __restrict__ Wkh,
    const _Float16* __restrict__ Wvh,
    _Float16* __restrict__ Qb, _Float16* __restrict__ Kb,
    _Float16* __restrict__ Vt) {
    const int lane = threadIdx.x & 31;
    const int wave = threadIdx.x >> 5;
    const int l16  = lane & 15;
    const int ksel = lane >> 4;
    const int strip = blockIdx.x * 4 + wave;   // 0..255 (32 tokens each)
    const int h     = blockIdx.y;
    const int which = blockIdx.z;              // 0=Q 1=K 2=V
    const _Float16* W = (which == 0) ? Wqh : (which == 1) ? Wkh : Wvh;
    const int mbase = strip * 32;
    const int nbase = h * DHH;

    v8f acc[2][4];
#pragma unroll
    for (int rt = 0; rt < 2; ++rt)
#pragma unroll
        for (int nt = 0; nt < 4; ++nt) acc[rt][nt] = fzero();

    for (int k = 0; k < DD; k += 32) {
        v16h a0 = fragA(xh, mbase + l16,      DD, k, ksel);
        v16h a1 = fragA(xh, mbase + 16 + l16, DD, k, ksel);
#pragma unroll
        for (int nt = 0; nt < 4; ++nt) {
            v16h b = fragB(W, nbase + nt * 16 + l16, DD, k, ksel);
            acc[0][nt] = wmma_f16(a0, b, acc[0][nt]);
            acc[1][nt] = wmma_f16(a1, b, acc[1][nt]);
        }
    }

    if (which < 2) {
        _Float16* dstbase = (which == 0) ? Qb : Kb;
        const float LN1E4 = 9.210340371976184f;       // ln(10000)
        const float inv0 = __expf(-(2.0f * (float)l16 / 64.0f) * LN1E4);
        const float inv1 = __expf(-(2.0f * (float)(l16 + 16) / 64.0f) * LN1E4);
#pragma unroll
        for (int rt = 0; rt < 2; ++rt) {
#pragma unroll
            for (int r = 0; r < 8; ++r) {
                int m = r + 8 * ksel;
                int g = mbase + rt * 16 + m;  // row in (B*T)
                int t = g & (TT - 1);
                int b = g >> 11;
                float a0 = (float)t * inv0, a1 = (float)t * inv1;
                float c0 = __cosf(a0), s0 = __sinf(a0);
                float c1 = __cosf(a1), s1 = __sinf(a1);
                float q0 = acc[rt][0][r], q1 = acc[rt][1][r];
                float q2 = acc[rt][2][r], q3 = acc[rt][3][r];
                float n0 = q0 * c0 - q2 * s0;
                float n2 = q2 * c0 + q0 * s0;
                float n1 = q1 * c1 - q3 * s1;
                float n3 = q3 * c1 + q1 * s1;
                _Float16* dst = dstbase + ((size_t)(b * HH + h) * TT + t) * DHH;
                dst[0 * 16 + l16] = (_Float16)n0;
                dst[1 * 16 + l16] = (_Float16)n1;
                dst[2 * 16 + l16] = (_Float16)n2;
                dst[3 * 16 + l16] = (_Float16)n3;
            }
        }
    } else {
        // V: store transposed  Vt[b,h,dh,t]
#pragma unroll
        for (int rt = 0; rt < 2; ++rt) {
#pragma unroll
            for (int r = 0; r < 8; ++r) {
                int m = r + 8 * ksel;
                int g = mbase + rt * 16 + m;
                int t = g & (TT - 1);
                int b = g >> 11;
                _Float16* dst = Vt + (size_t)(b * HH + h) * DHH * TT;
#pragma unroll
                for (int nt = 0; nt < 4; ++nt)
                    dst[(size_t)(nt * 16 + l16) * TT + t] = (_Float16)acc[rt][nt][r];
            }
        }
    }
}

// ---------------- flash attention ----------------
__device__ __forceinline__ void softmax_update(
    v8f& s0, v8f& s1, float (&mr)[8], float (&lr)[8], v8f (&o)[4],
    _Float16* ldsP, int qbase, int k0g, int l16, int ksel) {
#pragma unroll
    for (int r = 0; r < 8; ++r) {
        int m  = r + 8 * ksel;
        int mg = qbase + m;
        float v0 = s0[r] * 0.125f;
        float v1 = s1[r] * 0.125f;
        if (k0g + l16 > mg)      v0 = -1e30f;
        if (k0g + 16 + l16 > mg) v1 = -1e30f;
        float rm = fmaxf(v0, v1);
        rm = fmaxf(rm, __shfl_xor(rm, 1, 32));
        rm = fmaxf(rm, __shfl_xor(rm, 2, 32));
        rm = fmaxf(rm, __shfl_xor(rm, 4, 32));
        rm = fmaxf(rm, __shfl_xor(rm, 8, 32));
        float mn = fmaxf(mr[r], rm);
        float alpha = __expf(mr[r] - mn);
        float p0 = __expf(v0 - mn);
        float p1 = __expf(v1 - mn);
        float rs = p0 + p1;
        rs += __shfl_xor(rs, 1, 32);
        rs += __shfl_xor(rs, 2, 32);
        rs += __shfl_xor(rs, 4, 32);
        rs += __shfl_xor(rs, 8, 32);
        lr[r] = lr[r] * alpha + rs;
        mr[r] = mn;
        o[0][r] *= alpha; o[1][r] *= alpha; o[2][r] *= alpha; o[3][r] *= alpha;
        ldsP[m * 32 + l16]      = (_Float16)p0;
        ldsP[m * 32 + 16 + l16] = (_Float16)p1;
    }
}

// grid: 1024 blocks x 128 threads; one wave = one (b,h) x 32-query strip
__global__ __launch_bounds__(128) void attn_kernel(
    const _Float16* __restrict__ Qb, const _Float16* __restrict__ Kb,
    const _Float16* __restrict__ Vt, _Float16* __restrict__ Oh) {
    __shared__ __align__(16) _Float16 lds[4][32 * 32];
    const int lane = threadIdx.x & 31;
    const int wave = threadIdx.x >> 5;
    const int l16  = lane & 15;
    const int ksel = lane >> 4;
    const int idx = blockIdx.x * 4 + wave;     // 0..4095
    const int qs  = idx & 63;                  // 32-query strip (T/32 = 64)
    const int bh  = idx >> 6;                  // 0..63
    const _Float16* Qp = Qb + (size_t)bh * TT * DHH;
    const _Float16* Kp = Kb + (size_t)bh * TT * DHH;
    const _Float16* Vp = Vt + (size_t)bh * DHH * TT;
    _Float16* ldsA = &lds[wave][0];
    _Float16* ldsB = &lds[wave][16 * 32];
    const int qbase = qs * 32;

    v16h aq00 = fragA(Qp, qbase + l16,      DHH, 0,  ksel);
    v16h aq01 = fragA(Qp, qbase + l16,      DHH, 32, ksel);
    v16h aq10 = fragA(Qp, qbase + 16 + l16, DHH, 0,  ksel);
    v16h aq11 = fragA(Qp, qbase + 16 + l16, DHH, 32, ksel);

    v8f oA[4], oB[4];
    float mrA[8], lrA[8], mrB[8], lrB[8];
#pragma unroll
    for (int i = 0; i < 4; ++i) { oA[i] = fzero(); oB[i] = fzero(); }
#pragma unroll
    for (int r = 0; r < 8; ++r) {
        mrA[r] = -1e30f; lrA[r] = 0.0f;
        mrB[r] = -1e30f; lrB[r] = 0.0f;
    }

    const int nkv = qs + 1;
    for (int kv = 0; kv < nkv; ++kv) {
        const int k0g = kv * 32;
        v8f z = fzero();
        v16h kb0a = fragB(Kp, k0g + l16,      DHH, 0,  ksel);
        v16h kb0b = fragB(Kp, k0g + l16,      DHH, 32, ksel);
        v16h kb1a = fragB(Kp, k0g + 16 + l16, DHH, 0,  ksel);
        v16h kb1b = fragB(Kp, k0g + 16 + l16, DHH, 32, ksel);
        // q-tile A (rows qbase..qbase+15)
        {
            v8f s0 = wmma_f16(aq01, kb0b, wmma_f16(aq00, kb0a, z));
            v8f s1 = wmma_f16(aq01, kb1b, wmma_f16(aq00, kb1a, z));
            softmax_update(s0, s1, mrA, lrA, oA, ldsA, qbase, k0g, l16, ksel);
        }
        // q-tile B (rows qbase+16..qbase+31)
        {
            v8f s0 = wmma_f16(aq11, kb0b, wmma_f16(aq10, kb0a, z));
            v8f s1 = wmma_f16(aq11, kb1b, wmma_f16(aq10, kb1a, z));
            softmax_update(s0, s1, mrB, lrB, oB, ldsB, qbase + 16, k0g, l16, ksel);
        }
        asm volatile("s_wait_dscnt 0" ::: "memory");
        v16h apA = fragA(ldsA, l16, 32, 0, ksel);
        v16h apB = fragA(ldsB, l16, 32, 0, ksel);
#pragma unroll
        for (int nt = 0; nt < 4; ++nt) {
            v16h bv = fragB(Vp, nt * 16 + l16, TT, k0g, ksel);
            oA[nt] = wmma_f16(apA, bv, oA[nt]);
            oB[nt] = wmma_f16(apB, bv, oB[nt]);
        }
    }

    const int b = bh >> 4, h = bh & 15;
#pragma unroll
    for (int r = 0; r < 8; ++r) {
        int m = r + 8 * ksel;
        {
            float inv = 1.0f / lrA[r];
            _Float16* dst = Oh + ((size_t)(b * TT + qbase + m)) * DD + h * DHH;
#pragma unroll
            for (int nt = 0; nt < 4; ++nt)
                dst[nt * 16 + l16] = (_Float16)(oA[nt][r] * inv);
        }
        {
            float inv = 1.0f / lrB[r];
            _Float16* dst = Oh + ((size_t)(b * TT + qbase + 16 + m)) * DD + h * DHH;
#pragma unroll
            for (int nt = 0; nt < 4; ++nt)
                dst[nt * 16 + l16] = (_Float16)(oB[nt][r] * inv);
        }
    }
}

// ---------------- output projection  out = O @ Wo^T  (f32 out) ----------------
// grid: (64, 16), block 128. wave -> 32 rows x 64 cols
__global__ __launch_bounds__(128) void out_proj_kernel(
    const _Float16* __restrict__ Oh, const _Float16* __restrict__ Woh,
    float* __restrict__ out) {
    const int lane = threadIdx.x & 31;
    const int wave = threadIdx.x >> 5;
    const int l16  = lane & 15;
    const int ksel = lane >> 4;
    const int strip = blockIdx.x * 4 + wave;  // 0..255
    const int nb = blockIdx.y;                // 0..15
    const int mbase = strip * 32;
    const int nbase = nb * 64;

    v8f acc[2][4];
#pragma unroll
    for (int rt = 0; rt < 2; ++rt)
#pragma unroll
        for (int nt = 0; nt < 4; ++nt) acc[rt][nt] = fzero();

    for (int k = 0; k < DD; k += 32) {
        v16h a0 = fragA(Oh, mbase + l16,      DD, k, ksel);
        v16h a1 = fragA(Oh, mbase + 16 + l16, DD, k, ksel);
#pragma unroll
        for (int nt = 0; nt < 4; ++nt) {
            v16h b = fragB(Woh, nbase + nt * 16 + l16, DD, k, ksel);
            acc[0][nt] = wmma_f16(a0, b, acc[0][nt]);
            acc[1][nt] = wmma_f16(a1, b, acc[1][nt]);
        }
    }
#pragma unroll
    for (int rt = 0; rt < 2; ++rt) {
#pragma unroll
        for (int r = 0; r < 8; ++r) {
            int m = r + 8 * ksel;
            float* dst = out + (size_t)(mbase + rt * 16 + m) * DD + nbase;
#pragma unroll
            for (int nt = 0; nt < 4; ++nt) dst[nt * 16 + l16] = acc[rt][nt][r];
        }
    }
}

extern "C" void kernel_launch(void* const* d_in, const int* in_sizes, int n_in,
                              void* d_out, int out_size, void* d_ws, size_t ws_size,
                              hipStream_t stream) {
    const float* x  = (const float*)d_in[0];
    const float* Wq = (const float*)d_in[1];
    const float* Wk = (const float*)d_in[2];
    const float* Wv = (const float*)d_in[3];
    const float* Wo = (const float*)d_in[4];
    float* out = (float*)d_out;

    _Float16* ws = (_Float16*)d_ws;
    size_t off = 0;
    _Float16* xh  = ws + off; off += (size_t)BB * TT * DD;   // 8M halfs
    _Float16* Wqh = ws + off; off += (size_t)DD * DD;
    _Float16* Wkh = ws + off; off += (size_t)DD * DD;
    _Float16* Wvh = ws + off; off += (size_t)DD * DD;
    _Float16* Woh = ws + off; off += (size_t)DD * DD;
    _Float16* Qb  = ws + off; off += (size_t)BB * HH * TT * DHH;
    _Float16* Kb  = ws + off; off += (size_t)BB * HH * TT * DHH;
    _Float16* Vt  = ws + off; off += (size_t)BB * HH * TT * DHH;
    _Float16* Oh  = xh;   // xh is dead after qkv_proj; reuse for attention out

    const int nx = BB * TT * DD;
    const int nw = DD * DD;
    cvt_f16_kernel<<<(nx + 255) / 256, 256, 0, stream>>>(x,  xh,  nx);
    cvt_f16_kernel<<<(nw + 255) / 256, 256, 0, stream>>>(Wq, Wqh, nw);
    cvt_f16_kernel<<<(nw + 255) / 256, 256, 0, stream>>>(Wk, Wkh, nw);
    cvt_f16_kernel<<<(nw + 255) / 256, 256, 0, stream>>>(Wv, Wvh, nw);
    cvt_f16_kernel<<<(nw + 255) / 256, 256, 0, stream>>>(Wo, Woh, nw);

    qkv_proj_kernel<<<dim3(64, 16, 3), 128, 0, stream>>>(xh, Wqh, Wkh, Wvh,
                                                         Qb, Kb, Vt);
    attn_kernel<<<dim3(1024), 128, 0, stream>>>(Qb, Kb, Vt, Oh);
    out_proj_kernel<<<dim3(64, 16), 128, 0, stream>>>(Oh, Woh, out);
}
